// TransformerEncoderLayer_47253230190700
// MI455X (gfx1250) — matrix operand order repaired and checked by reading, not compiled, not run on hardware
//
#include <hip/hip_runtime.h>

// ---------------------------------------------------------------------------
// Types for CDNA5 WMMA (wave32): v_wmma_f32_16x16x32_bf16
// ---------------------------------------------------------------------------
typedef __bf16 bf16_t;
typedef __attribute__((ext_vector_type(16))) __bf16 v16bf;
typedef __attribute__((ext_vector_type(8)))  __bf16 v8bf;
typedef __attribute__((ext_vector_type(2)))  __bf16 v2bf;
typedef __attribute__((ext_vector_type(8)))  float  v8f;
typedef __attribute__((ext_vector_type(4)))  int    v4i;

// ---------------------------------------------------------------------------
// Async global->LDS (GLOBAL_LOAD_ASYNC_TO_LDS_B128, ASYNCcnt-tracked).
// Builtin params are typed: (v4i AS1*, v4i AS3*, imm offset, imm cpol).
// Guarded: falls back to plain LDS staging if the builtin is absent.
// ---------------------------------------------------------------------------
#if __has_builtin(__builtin_amdgcn_global_load_async_to_lds_b128)
#define USE_ASYNC 1
typedef __attribute__((address_space(1))) v4i gv4i;  // global int4
typedef __attribute__((address_space(3))) v4i lv4i;  // LDS int4
#else
#define USE_ASYNC 0
#endif

__device__ inline void copy16_g2l(const bf16_t* gsrc, bf16_t* ldst) {
#if USE_ASYNC
  __builtin_amdgcn_global_load_async_to_lds_b128((gv4i*)gsrc, (lv4i*)ldst, 0, 0);
#else
  *(v8bf*)ldst = *(const v8bf*)gsrc;
#endif
}

__device__ inline void wait_async0() {
#if USE_ASYNC
#if __has_builtin(__builtin_amdgcn_s_wait_asynccnt)
  __builtin_amdgcn_s_wait_asynccnt(0);
#else
  asm volatile("s_wait_asynccnt 0" ::: "memory");
#endif
#endif
}

// ---------------------------------------------------------------------------
// Fragment loaders from LDS, matching the CDNA5 ISA 16-bit layouts.
//
// A (16x32, MxK): lane l(0..15) row M=l; half hi=lane>>4:
//   elems 0..7  = K[k0 + hi*8 .. +7]
//   elems 8..15 = K[k0 + hi*8 + 16 .. +23]
// B (32x16, KxN) staged TRANSPOSED in LDS as BT[n][k] so that lane (hi,l)
//   reads the same K order as A: elems 0..7 = K[k0 + hi*8 ..], 8..15 = K[k0 + hi*8 + 16 ..].
// Both loads are 16B-aligned -> ds_load_b128.
// ---------------------------------------------------------------------------
__device__ inline v16bf load_a_frag(const bf16_t* sm, int ld, int row0, int k0) {
  int lane = threadIdx.x & 31;
  int l = lane & 15, hi = lane >> 4;
  const bf16_t* p = sm + (size_t)(row0 + l) * ld + k0 + hi * 8;
  v16bf a;
  ((v8bf*)&a)[0] = *(const v8bf*)p;
  ((v8bf*)&a)[1] = *(const v8bf*)(p + 16);
  return a;
}

__device__ inline v16bf load_bt_frag(const bf16_t* sm, int ld, int n0, int k0) {
  int lane = threadIdx.x & 31;
  int l = lane & 15, hi = lane >> 4;
  const bf16_t* p = sm + (size_t)(n0 + l) * ld + k0 + hi * 8;
  v16bf b;
  ((v8bf*)&b)[0] = *(const v8bf*)p;
  ((v8bf*)&b)[1] = *(const v8bf*)(p + 16);
  return b;
}

__device__ inline v8f wmma_bf16(v16bf a, v16bf b, v8f c) {
  return __builtin_amdgcn_wmma_f32_16x16x32_bf16(false, a, false, b,
                                                 (short)0, c, false, false);
}

// ---------------------------------------------------------------------------
// fp32 -> bf16 convert (weights)
// ---------------------------------------------------------------------------
__global__ __launch_bounds__(256) void cvt_f32_bf16(const float* __restrict__ src,
                                                    bf16_t* __restrict__ dst, int n) {
  int i = blockIdx.x * 256 + threadIdx.x;
  int stride = gridDim.x * 256;
  for (; i < n; i += stride) dst[i] = (bf16_t)src[i];
}

// ---------------------------------------------------------------------------
// LayerNorm: one row (E=1024) per block, bf16 output
// ---------------------------------------------------------------------------
__global__ __launch_bounds__(256) void ln_bf16(const float* __restrict__ x,
                                               const float* __restrict__ g,
                                               const float* __restrict__ b,
                                               bf16_t* __restrict__ out, int E) {
  __shared__ float red[256];
  __shared__ float red2[256];
  int row = blockIdx.x;
  int tid = threadIdx.x;
  const float* xr = x + (size_t)row * E;
  float s = 0.f, s2 = 0.f;
  for (int c = tid; c < E; c += 256) { float v = xr[c]; s += v; s2 += v * v; }
  red[tid] = s; red2[tid] = s2;
  __syncthreads();
  for (int off = 128; off > 0; off >>= 1) {
    if (tid < off) { red[tid] += red[tid + off]; red2[tid] += red2[tid + off]; }
    __syncthreads();
  }
  float mu = red[0] / (float)E;
  float var = red2[0] / (float)E - mu * mu;
  float rs = rsqrtf(var + 1e-5f);
  bf16_t* outr = out + (size_t)row * E;
  for (int c = tid; c < E; c += 256)
    outr[c] = (bf16_t)((xr[c] - mu) * rs * g[c] + b[c]);
}

// ---------------------------------------------------------------------------
// Generic bf16 GEMM: C = A[MxK] * B[KxN] + bias, epilogue by mode.
//   mode 0: QKV -> bf16 dst in [B,H,N,D] layout, val *= scale
//   mode 1: fp32 dst = val + resid   (out-proj / FFN2, residual fused)
//   mode 2: bf16 dst = relu(val)     (FFN1)
// Block: 256 thr (8 waves). Tile: BM=64, BN=128, BK=32.
// Wave grid 2(M) x 4(N); each wave: 32x32 = 2x2 WMMA tiles (frag reuse).
// A tile staged via async global->LDS; B tile transposed with paired-k
// 32-bit LDS stores (half the DS-store count of 16-bit scatter).
// ---------------------------------------------------------------------------
#define BM 64
#define BN 128
#define BK 32
#define ALD 40   // BK + 8 pad (keeps 16B alignment: 40*2=80B)
#define BLD 40

__global__ __launch_bounds__(256) void gemm_bf16(
    const bf16_t* __restrict__ A, const bf16_t* __restrict__ Bm,
    const float* __restrict__ bias, int M, int K, int N,
    int mode, float scale, void* __restrict__ dst,
    const float* __restrict__ resid) {
  __shared__ bf16_t As[BM][ALD];
  __shared__ bf16_t Bs[BN][BLD];

  int tid = threadIdx.x;
  int row0 = blockIdx.y * BM;
  int n0 = blockIdx.x * BN;
  int w = tid >> 5;
  int wm = w & 1;        // 0..1 over M
  int wn = w >> 1;       // 0..3 over N
  int lane = tid & 31, l = lane & 15, hi = lane >> 4;

  v8f acc[2][2] = {};

  int ar = tid >> 2, ac = (tid & 3) * 8;          // A stage: 64 x 32, 8 elem/thr
  int bkr = (tid >> 4) * 2, bnb = (tid & 15) * 8; // B stage: 2 k-rows x 8 n/thr

  for (int k0 = 0; k0 < K; k0 += BK) {
    // ---- stage A tile (row-major) via async global->LDS ----
    copy16_g2l(A + (size_t)(row0 + ar) * K + k0 + ac, &As[ar][ac]);
    // ---- stage B tile transposed: Bs[n][k], paired-k b32 stores ----
    {
      const bf16_t* gp = Bm + (size_t)(k0 + bkr) * N + n0 + bnb;
      v8bf b0 = *(const v8bf*)gp;
      v8bf b1 = *(const v8bf*)(gp + N);
#pragma unroll
      for (int i = 0; i < 8; ++i) {
        v2bf t; t[0] = b0[i]; t[1] = b1[i];
        *(v2bf*)&Bs[bnb + i][bkr] = t;
      }
    }
    // prefetch next B tile into cache (global_prefetch on gfx1250)
    if (k0 + BK < K)
      __builtin_prefetch(Bm + (size_t)(k0 + BK + bkr) * N + n0 + bnb, 0, 1);
    wait_async0();
    __syncthreads();

    v16bf a0 = load_a_frag(&As[0][0], ALD, wm * 32, 0);
    v16bf a1 = load_a_frag(&As[0][0], ALD, wm * 32 + 16, 0);
    v16bf b0 = load_bt_frag(&Bs[0][0], BLD, wn * 32, 0);
    v16bf b1 = load_bt_frag(&Bs[0][0], BLD, wn * 32 + 16, 0);
    acc[0][0] = wmma_bf16(a0, b0, acc[0][0]);
    acc[0][1] = wmma_bf16(a0, b1, acc[0][1]);
    acc[1][0] = wmma_bf16(a1, b0, acc[1][0]);
    acc[1][1] = wmma_bf16(a1, b1, acc[1][1]);
    __syncthreads();
  }

  // ---- epilogue ----
#pragma unroll
  for (int ti = 0; ti < 2; ++ti)
#pragma unroll
    for (int tj = 0; tj < 2; ++tj)
#pragma unroll
      for (int r = 0; r < 8; ++r) {
        int row = row0 + wm * 32 + ti * 16 + hi * 8 + r;
        int col = n0 + wn * 32 + tj * 16 + l;
        float val = acc[ti][tj][r];
        if (bias) val += bias[col];
        val *= scale;
        if (mode == 0) {
          int bb = row >> 10, nn = row & 1023;
          int hh = col >> 6, dd = col & 63;
          ((bf16_t*)dst)[(((size_t)bb * 16 + hh) * 1024 + nn) * 64 + dd] = (bf16_t)val;
        } else if (mode == 1) {
          size_t idx = (size_t)row * N + col;
          ((float*)dst)[idx] = val + resid[idx];
        } else {
          ((bf16_t*)dst)[(size_t)row * N + col] = (bf16_t)fmaxf(val, 0.0f);
        }
      }
}

// ---------------------------------------------------------------------------
// Attention: grid (N/32, B*H), block 256 (8 waves).
// q,k,v bf16 in [B,H,N,64]; q pre-scaled by 1/8. ctx bf16 out [B,N,E].
// Whole 32x1024 score block lives in LDS (CDNA5: 320KB/WGP); flash-style
// un-normalized softmax, divide by row-sum at the end.
// Q/K blocks staged via async global->LDS; V^T staged with paired-key b32.
// Dynamic LDS layout (bytes):
//   Ss fp32 [32][1032]   @ 0       (132096)
//   Ps bf16 [32][1032]   @ 132096  (66048)
//   Qs bf16 [32][72]     @ 198144  (4608)
//   KVs bf16 [64][72]    @ 202752  (9216)   (K block, then V^T block)
//   red fp32 [32][8]     @ 211968  (1024)
//   rowsum fp32 [32]     @ 212992  (128)    total 213120
// ---------------------------------------------------------------------------
#define ATTN_LDS_BYTES 213120

__global__ __launch_bounds__(256) void attn_bf16(
    const bf16_t* __restrict__ q, const bf16_t* __restrict__ k,
    const bf16_t* __restrict__ v, bf16_t* __restrict__ ctx) {
  extern __shared__ char smem_raw[];
  float*  Ss     = (float*)smem_raw;                 // [32][1032]
  bf16_t* Ps     = (bf16_t*)(smem_raw + 132096);     // [32][1032]
  bf16_t* Qs     = (bf16_t*)(smem_raw + 198144);     // [32][72]
  bf16_t* KVs    = (bf16_t*)(smem_raw + 202752);     // [64][72]
  float*  red    = (float*)(smem_raw + 211968);      // [32][8]
  float*  rowsum = (float*)(smem_raw + 212992);      // [32]

  int tid = threadIdx.x;
  int bh = blockIdx.y;                 // b*16 + h
  int qbase = blockIdx.x * 32;
  const bf16_t* Qg = q + ((size_t)bh * 1024 + qbase) * 64;
  const bf16_t* Kg = k + (size_t)bh * 1024 * 64;
  const bf16_t* Vg = v + (size_t)bh * 1024 * 64;

  int w = tid >> 5, wm = w & 1, wn = w >> 1;   // 2(M) x 4(N) wave grid
  int lane = tid & 31, l = lane & 15, hi = lane >> 4;

  // stage Q block 32x64 (async, 8 elem/thr)
  {
    int r = tid >> 3, c = (tid & 7) * 8;
    copy16_g2l(Qg + (size_t)r * 64 + c, Qs + r * 72 + c);
  }
  wait_async0();
  __syncthreads();

  // ---- Phase 1: S = Q * K^T, one 64-key block at a time ----
  for (int kb = 0; kb < 16; ++kb) {
    {
      int r = tid >> 2, c = (tid & 3) * 16;      // 64x64, 16 elem/thr (async)
      const bf16_t* gp = Kg + (size_t)(kb * 64 + r) * 64 + c;
      copy16_g2l(gp,     KVs + r * 72 + c);
      copy16_g2l(gp + 8, KVs + r * 72 + c + 8);
    }
    wait_async0();
    __syncthreads();
    v8f sa = {};
#pragma unroll
    for (int ks = 0; ks < 64; ks += 32) {
      v16bf af = load_a_frag(Qs, 72, wm * 16, ks);
      v16bf bt = load_bt_frag(KVs, 72, wn * 16, ks);  // BT[n=key][k=d] = K[key][d]
      sa = wmma_bf16(af, bt, sa);
    }
#pragma unroll
    for (int r = 0; r < 8; ++r)
      Ss[(size_t)(wm * 16 + hi * 8 + r) * 1032 + kb * 64 + wn * 16 + l] = sa[r];
    __syncthreads();
  }

  // ---- Phase 2: row softmax (un-normalized), P in bf16, keep row sums ----
  {
    int r = tid >> 3, part = tid & 7;
    int c0 = part * 128;
    float m = -3.4e38f;
    for (int c = 0; c < 128; ++c) m = fmaxf(m, Ss[(size_t)r * 1032 + c0 + c]);
    red[r * 8 + part] = m;
    __syncthreads();
    float mm = red[r * 8];
    for (int i = 1; i < 8; ++i) mm = fmaxf(mm, red[r * 8 + i]);
    float s = 0.f;
    for (int c = 0; c < 128; ++c) {
      float e = __expf(Ss[(size_t)r * 1032 + c0 + c] - mm);
      Ps[(size_t)r * 1032 + c0 + c] = (bf16_t)e;
      s += e;
    }
    __syncthreads();                   // all max reads done before reuse of red
    red[r * 8 + part] = s;
    __syncthreads();
    if (part == 0) {
      float t = 0.f;
      for (int i = 0; i < 8; ++i) t += red[r * 8 + i];
      rowsum[r] = t;
    }
  }
  __syncthreads();

  // ---- Phase 3: O = P * V ----
  v8f oacc = {};
  for (int kb = 0; kb < 16; ++kb) {
    {
      // stage V^T: KVs[d][key], paired-key b32 stores
      int kp = (tid >> 3) * 2, db = (tid & 7) * 8;   // 2 keys x 8 d per thread
      const bf16_t* gp = Vg + (size_t)(kb * 64 + kp) * 64 + db;
      v8bf v0 = *(const v8bf*)gp;
      v8bf v1 = *(const v8bf*)(gp + 64);
#pragma unroll
      for (int i = 0; i < 8; ++i) {
        v2bf t; t[0] = v0[i]; t[1] = v1[i];
        *(v2bf*)&KVs[(db + i) * 72 + kp] = t;
      }
    }
    __syncthreads();
#pragma unroll
    for (int ks = 0; ks < 64; ks += 32) {
      v16bf af = load_a_frag(Ps, 1032, wm * 16, kb * 64 + ks);
      v16bf bt = load_bt_frag(KVs, 72, wn * 16, ks);   // BT[n=d][k=key] = V[key][d]
      oacc = wmma_bf16(af, bt, oacc);
    }
    __syncthreads();
  }

  // ---- epilogue: ctx[b][n][h*64+d] = O / rowsum ----
  int b = bh >> 4, h = bh & 15;
#pragma unroll
  for (int r = 0; r < 8; ++r) {
    int row = wm * 16 + hi * 8 + r;
    int d = wn * 16 + l;
    float val = oacc[r] / rowsum[row];
    ctx[((size_t)b * 1024 + qbase + row) * 1024 + h * 64 + d] = (bf16_t)val;
  }
}

// ---------------------------------------------------------------------------
// Launch. Inputs (fp32): x, ln1_g, ln1_b, ln2_g, ln2_b, wq, bq, wk, bk,
//                        wv, bv, wo, bo, w1, b1, w2, b2
// Workspace layout (bf16 unless noted), total 184 MB:
//   wq 0 | wk 2M | wv 4M | wo 6M | w1 8M | w2 16M | h 24M | q 40M | k 56M
//   v 72M | ctx 88M | h2 104M | ff1 120M..184M
// ---------------------------------------------------------------------------
extern "C" void kernel_launch(void* const* d_in, const int* in_sizes, int n_in,
                              void* d_out, int out_size, void* d_ws, size_t ws_size,
                              hipStream_t stream) {
  const float* x     = (const float*)d_in[0];
  const float* ln1_g = (const float*)d_in[1];
  const float* ln1_b = (const float*)d_in[2];
  const float* ln2_g = (const float*)d_in[3];
  const float* ln2_b = (const float*)d_in[4];
  const float* wq = (const float*)d_in[5];
  const float* bq = (const float*)d_in[6];
  const float* wk = (const float*)d_in[7];
  const float* bk = (const float*)d_in[8];
  const float* wv = (const float*)d_in[9];
  const float* bv = (const float*)d_in[10];
  const float* wo = (const float*)d_in[11];
  const float* bo = (const float*)d_in[12];
  const float* w1 = (const float*)d_in[13];
  const float* b1 = (const float*)d_in[14];
  const float* w2 = (const float*)d_in[15];
  const float* b2 = (const float*)d_in[16];
  float* out = (float*)d_out;

  const int R = 8192, E = 1024, HID = 4096;
  const size_t MB = 1024ull * 1024ull;
  char* ws = (char*)d_ws;
  bf16_t* wqB  = (bf16_t*)(ws + 0 * MB);
  bf16_t* wkB  = (bf16_t*)(ws + 2 * MB);
  bf16_t* wvB  = (bf16_t*)(ws + 4 * MB);
  bf16_t* woB  = (bf16_t*)(ws + 6 * MB);
  bf16_t* w1B  = (bf16_t*)(ws + 8 * MB);
  bf16_t* w2B  = (bf16_t*)(ws + 16 * MB);
  bf16_t* hB   = (bf16_t*)(ws + 24 * MB);
  bf16_t* qB   = (bf16_t*)(ws + 40 * MB);
  bf16_t* kB   = (bf16_t*)(ws + 56 * MB);
  bf16_t* vB   = (bf16_t*)(ws + 72 * MB);
  bf16_t* ctxB = (bf16_t*)(ws + 88 * MB);
  bf16_t* h2B  = (bf16_t*)(ws + 104 * MB);
  bf16_t* ff1B = (bf16_t*)(ws + 120 * MB);

  // weights -> bf16
  cvt_f32_bf16<<<512, 256, 0, stream>>>(wq, wqB, E * E);
  cvt_f32_bf16<<<512, 256, 0, stream>>>(wk, wkB, E * E);
  cvt_f32_bf16<<<512, 256, 0, stream>>>(wv, wvB, E * E);
  cvt_f32_bf16<<<512, 256, 0, stream>>>(wo, woB, E * E);
  cvt_f32_bf16<<<1024, 256, 0, stream>>>(w1, w1B, E * HID);
  cvt_f32_bf16<<<1024, 256, 0, stream>>>(w2, w2B, HID * E);

  // LN1 -> h (bf16)
  ln_bf16<<<R, 256, 0, stream>>>(x, ln1_g, ln1_b, hB, E);

  // QKV projections -> [B,H,N,D] bf16 (q scaled by 1/sqrt(64))
  dim3 gE(E / BN, R / BM);
  gemm_bf16<<<gE, 256, 0, stream>>>(hB, wqB, bq, R, E, E, 0, 0.125f, qB, nullptr);
  gemm_bf16<<<gE, 256, 0, stream>>>(hB, wkB, bk, R, E, E, 0, 1.0f, kB, nullptr);
  gemm_bf16<<<gE, 256, 0, stream>>>(hB, wvB, bv, R, E, E, 0, 1.0f, vB, nullptr);

  // attention -> ctx (bf16 [B,N,E])
  dim3 gA(1024 / 32, 8 * 16);
  attn_bf16<<<gA, 256, ATTN_LDS_BYTES, stream>>>(qB, kB, vB, ctxB);

  // out-proj + residual -> d_out (fp32)  : x1 = x + ctx@wo + bo
  gemm_bf16<<<gE, 256, 0, stream>>>(ctxB, woB, bo, R, E, E, 1, 1.0f, out, x);

  // LN2 -> h2 (bf16)
  ln_bf16<<<R, 256, 0, stream>>>(out, ln2_g, ln2_b, h2B, E);

  // FFN1: relu(h2@w1 + b1) -> ff1 (bf16)
  dim3 gH(HID / BN, R / BM);
  gemm_bf16<<<gH, 256, 0, stream>>>(h2B, w1B, b1, R, E, HID, 2, 1.0f, ff1B, nullptr);

  // FFN2 + residual: out = x1 + ff1@w2 + b2
  gemm_bf16<<<gE, 256, 0, stream>>>(ff1B, w2B, b2, R, HID, E, 1, 1.0f, out, out);
}
